// WincBlock_60945585930338
// MI455X (gfx1250) — matrix operand order repaired and checked by reading, not compiled, not run on hardware
//
#include <hip/hip_runtime.h>
#include <math.h>

typedef _Float16 half_t;
typedef __attribute__((ext_vector_type(16))) _Float16 v16h;
typedef __attribute__((ext_vector_type(8)))  float    v8f;
typedef unsigned int u32;

#define LANE (threadIdx.x & 31)
#define WAVE ((int)(threadIdx.x >> 5))
#define NT_  256   // threads per block
#define NWV_ 8     // waves per block

// ---------------------------------------------------------------------------
// WMMA wrapper: D = A(16x32 f16) * B(32x16 f16) + C(16x16 f32)
// ---------------------------------------------------------------------------
__device__ __forceinline__ v8f wmma_f16(v16h a, v16h b, v8f c) {
  return __builtin_amdgcn_wmma_f32_16x16x32_f16(
      false, a, false, b, (short)0, c, false, false);
}

// A-matrix K index for half-slot i, lane-group lg (contiguous: k = 16*lg + i, matches B loaders)
__device__ __forceinline__ int a_kk(int i, int lg) {
  return (lg << 4) + i;
}

__device__ __forceinline__ v16h ldA_h(const half_t* src, int ld, int row0, int k0, int kmax) {
  int lg = LANE >> 4, lr = LANE & 15;
  const half_t* rp = src + (row0 + lr) * ld;
  v16h a;
#pragma unroll
  for (int i = 0; i < 16; ++i) {
    int K = k0 + a_kk(i, lg);
    a[i] = (K < kmax) ? rp[K] : (half_t)0.f;
  }
  return a;
}

__device__ __forceinline__ v16h ldA_f(const float* src, int ld, int row0, int k0, int kmax) {
  int lg = LANE >> 4, lr = LANE & 15;
  const float* rp = src + (row0 + lr) * ld;
  v16h a;
#pragma unroll
  for (int i = 0; i < 16; ++i) {
    int K = k0 + a_kk(i, lg);
    a[i] = (K < kmax) ? (half_t)rp[K] : (half_t)0.f;
  }
  return a;
}

// B fragment, element(K,n) = src[(col0+n)*ld + K] (B = srcT): per-lane contiguous runs
__device__ __forceinline__ v16h ldBt_h(const half_t* src, int ld, int k0, int col0, int kmax) {
  int lg = LANE >> 4, n = LANE & 15;
  const half_t* cp = src + (col0 + n) * ld;
  v16h b;
#pragma unroll
  for (int i = 0; i < 16; ++i) {
    int K = k0 + (lg << 4) + i;
    b[i] = (K < kmax) ? cp[K] : (half_t)0.f;
  }
  return b;
}

// B fragment, element(K,n) = src[K*ld + col0 + n] (row-major B)
__device__ __forceinline__ v16h ldB_h(const half_t* src, int ld, int k0, int col0, int kmax) {
  int lg = LANE >> 4, n = LANE & 15;
  v16h b;
#pragma unroll
  for (int i = 0; i < 16; ++i) {
    int K = k0 + (lg << 4) + i;
    b[i] = (K < kmax) ? src[K * ld + col0 + n] : (half_t)0.f;
  }
  return b;
}

// ---------------------------------------------------------------------------
#define B_   8
#define C_   96
#define H_   192
#define W_   192
#define WS_  8
#define N_   64
#define NH_  24
#define QD_  16
#define INNER_ 64
#define O_   96
#define SCALE_ 0.25f

// ws layout (bytes)
#define WS_ABIAS   0                       // 4096 f32
#define WS_QKVW    16384                   // 192*96 f16
#define WS_OUTW    53248                   // 96*64 f16
#define WS_W1      65536                   // 96*96 f16
#define WS_W2R     83968                   // [3][3][96][96] f16 (transposed conv2_w)
#define WS_XATTN   249856                  // 8*96*192*192 f32

// attn LDS layout
#define SM_SC     0                        // f32[4][64][64]   65536
#define SM_XH     65536                    // f16[64][96]      12288
#define SM_QKVH   (65536+12288)            // f16[64][192]     24576
#define SM_OH     (65536+12288+24576)      // f16[64][64]       8192
#define SM_XF     (SM_OH+8192)             // f32[96][64]      24576  (channel-major window)
#define SM_OT     (SM_XF+24576)            // f32[64][96]      24576
#define SM_WQ     (SM_OT+24576)            // f16[192][96]     36864
#define SM_WO     (SM_WQ+36864)            // f16[96][64]      12288
#define ATTN_SMEM (SM_WO+12288)            // 208896

// conv LDS layout
#define SM_XT     0                        // f16[336][96]     64512
#define SM_HT     64512                    // f16[336][96]     64512
#define SM_RT     129024                   // f32[256][96]     98304
#define SM_WS1    227328                   // f16[96][96]      18432
#define CONV_SMEM (SM_WS1+18432)           // 245760

// ---------------------------------------------------------------------------
// Kernel 0: attn-bias GEMV + weight conversion / transpose
// ---------------------------------------------------------------------------
__global__ void prep_kernel(const float* __restrict__ bias, const float* __restrict__ bpw,
                            const float* __restrict__ qkvw, const float* __restrict__ outw,
                            const float* __restrict__ w1, const float* __restrict__ w2,
                            float* __restrict__ abias, half_t* __restrict__ qkvwh,
                            half_t* __restrict__ outwh, half_t* __restrict__ w1h,
                            half_t* __restrict__ w2r) {
  int tid = blockIdx.x * blockDim.x + threadIdx.x;
  int stride = gridDim.x * blockDim.x;
  for (int i = tid; i < N_ * N_; i += stride) {
    float s = 0.f;
#pragma unroll
    for (int j = 0; j < N_; ++j) s += bias[j] * bpw[i * N_ + j];
    abias[i] = s;
  }
  for (int i = tid; i < 3 * INNER_ * C_; i += stride) qkvwh[i] = (half_t)qkvw[i];
  for (int i = tid; i < C_ * INNER_;     i += stride) outwh[i] = (half_t)outw[i];
  for (int i = tid; i < O_ * C_;         i += stride) w1h[i]   = (half_t)w1[i];
  // conv2_w (O,I,3,3) -> w2r[((dy*3+dx)*O + o)*O + i]
  for (int i = tid; i < O_ * O_ * 9; i += stride) {
    int dx = i % 3, t1 = i / 3;
    int dy = t1 % 3, t2 = t1 / 3;
    int ci = t2 % O_, o = t2 / O_;
    w2r[((dy * 3 + dx) * O_ + o) * O_ + ci] = (half_t)w2[i];
  }
}

// ---------------------------------------------------------------------------
// Kernel 1: window MHA, 1 block (8 waves) per window. Writes xattn = x + o.
// ---------------------------------------------------------------------------
__global__ void attn_kernel(const float* __restrict__ x, const float* __restrict__ abias,
                            const half_t* __restrict__ qkvwh, const half_t* __restrict__ outwh,
                            const float* __restrict__ qkvb, const float* __restrict__ outb,
                            float* __restrict__ xattn) {
  extern __shared__ char smem[];
  float*  sc   = (float*)(smem + SM_SC);
  half_t* xh   = (half_t*)(smem + SM_XH);
  half_t* qkvh = (half_t*)(smem + SM_QKVH);
  half_t* oh   = (half_t*)(smem + SM_OH);
  float*  xf   = (float*)(smem + SM_XF);    // [c][t] channel-major window, f32
  float*  ot   = (float*)(smem + SM_OT);    // [t][c] out-proj result, f32
  half_t* wq   = (half_t*)(smem + SM_WQ);
  half_t* wo   = (half_t*)(smem + SM_WO);

  const int win = blockIdx.x;
  const int b   = win / (NH_ * NH_);
  const int wr  = win % (NH_ * NH_);
  const int wh  = wr / NH_;
  const int ww  = wr % NH_;
  const int tid = threadIdx.x;
  const int lg = LANE >> 4, ln = LANE & 15;

  // ---- async-stage the f32 window into LDS (channel-major; 16B runs) ----
  // global run: x[b][c][wh*8+r][ww*8 .. +7] is 32B contiguous -> 2 x b128
  for (int j = tid; j < C_ * WS_ * 2; j += NT_) {
    int c = j >> 4, r = (j >> 1) & 7, hf = j & 1;
    const float* gp = x + ((b * C_ + c) * H_ + wh * WS_ + r) * W_ + ww * WS_ + hf * 4;
    u32 lds = (u32)(size_t)(const void*)&xf[c * N_ + r * 8 + hf * 4];
    asm volatile("global_load_async_to_lds_b128 %0, %1, off" :: "v"(lds), "v"(gp) : "memory");
  }
  // ---- stage weights into LDS (contiguous 16B chunks) ----
  for (int j = tid; j < (3 * INNER_ * C_) / 8; j += NT_)
    ((uint4*)wq)[j] = ((const uint4*)qkvwh)[j];
  for (int j = tid; j < (C_ * INNER_) / 8; j += NT_)
    ((uint4*)wo)[j] = ((const uint4*)outwh)[j];
  asm volatile("s_wait_asynccnt 0x0" ::: "memory");
  __syncthreads();

  // ---- convert window to f16 token-major [t][c] ----
  for (int idx = tid; idx < N_ * C_; idx += NT_) {
    int c = idx / N_, t = idx % N_;
    xh[t * C_ + c] = (half_t)xf[c * N_ + t];
  }
  __syncthreads();

  // ---- qkv = xw @ qkv_w.T + qkv_b : 48 tiles over 8 waves ----
  for (int tile = WAVE; tile < 4 * 12; tile += NWV_) {
    int mt = tile / 12, nt = tile % 12;
    v8f acc = {};
#pragma unroll
    for (int k = 0; k < 3; ++k) {
      v16h a  = ldA_h(xh, C_, mt * 16, k * 32, C_);
      v16h bm = ldBt_h(wq, C_, k * 32, nt * 16, C_);
      acc = wmma_f16(a, bm, acc);
    }
    float bb = qkvb[nt * 16 + ln];
#pragma unroll
    for (int v = 0; v < 8; ++v)
      qkvh[(mt * 16 + lg * 8 + v) * 192 + nt * 16 + ln] = (half_t)(acc[v] + bb);
  }
  __syncthreads();

  // ---- scores = q k^T * SCALE + bias : 64 tiles (head,mt,nt) ----
  for (int tile = WAVE; tile < 64; tile += NWV_) {
    int head = tile >> 4, mt = (tile >> 2) & 3, nt = tile & 3;
    float* S = sc + head * N_ * N_;
    int qo = head * QD_, ko = INNER_ + head * QD_;
    v8f acc = {};
    v16h a  = ldA_h(qkvh, 192, mt * 16, qo, qo + QD_);
    v16h bm = ldBt_h(qkvh, 192, ko, nt * 16, ko + QD_);
    acc = wmma_f16(a, bm, acc);
#pragma unroll
    for (int v = 0; v < 8; ++v) {
      int r = mt * 16 + lg * 8 + v, c = nt * 16 + ln;
      S[r * N_ + c] = acc[v] * SCALE_ + abias[r * N_ + c];
    }
  }
  __syncthreads();

  // ---- softmax: one row per thread (256 rows total) ----
  {
    float* row = sc + tid * N_;  // tid = head*64 + r
    float mx = -1e30f;
    for (int j = 0; j < N_; ++j) mx = fmaxf(mx, row[j]);
    float sum = 0.f;
    for (int j = 0; j < N_; ++j) { float e = __expf(row[j] - mx); row[j] = e; sum += e; }
    float inv = 1.f / sum;
    for (int j = 0; j < N_; ++j) row[j] *= inv;
  }
  __syncthreads();

  // ---- o = attn @ v : 16 tiles (head, mt) ----
  for (int tile = WAVE; tile < 16; tile += NWV_) {
    int head = tile >> 2, mt = tile & 3;
    const float* S = sc + head * N_ * N_;
    int vo = 2 * INNER_ + head * QD_;
    v8f acc = {};
#pragma unroll
    for (int kt = 0; kt < 2; ++kt) {
      v16h a  = ldA_f(S, N_, mt * 16, kt * 32, N_);
      v16h bm = ldB_h(qkvh, 192, kt * 32, vo, N_);
      acc = wmma_f16(a, bm, acc);
    }
#pragma unroll
    for (int v = 0; v < 8; ++v)
      oh[(mt * 16 + lg * 8 + v) * INNER_ + head * QD_ + ln] = (half_t)acc[v];
  }
  __syncthreads();

  // ---- out proj -> LDS ot: 24 tiles ----
  for (int tile = WAVE; tile < 4 * 6; tile += NWV_) {
    int mt = tile / 6, nt = tile % 6;
    v8f acc = {};
#pragma unroll
    for (int kt = 0; kt < 2; ++kt) {
      v16h a  = ldA_h(oh, INNER_, mt * 16, kt * 32, INNER_);
      v16h bm = ldBt_h(wo, INNER_, kt * 32, nt * 16, INNER_);
      acc = wmma_f16(a, bm, acc);
    }
    int c = nt * 16 + ln;
    float bb = outb[c];
#pragma unroll
    for (int v = 0; v < 8; ++v)
      ot[(mt * 16 + lg * 8 + v) * C_ + c] = acc[v] + bb;
  }
  __syncthreads();

  // ---- coalesced residual writeback: xattn = x + o ----
  for (int idx = tid; idx < N_ * C_; idx += NT_) {
    int c = idx / N_, t = idx % N_;
    int hh = wh * WS_ + (t >> 3), wp = ww * WS_ + (t & 7);
    xattn[((b * C_ + c) * H_ + hh) * W_ + wp] = xf[c * N_ + t] + ot[t * C_ + c];
  }
}

// ---------------------------------------------------------------------------
// Kernel 2: fused conv MLP, 1 block (8 waves) per 16x16 output tile.
// ---------------------------------------------------------------------------
__global__ void conv_kernel(const float* __restrict__ xattn, const half_t* __restrict__ w1h,
                            const half_t* __restrict__ w2r, const float* __restrict__ b1,
                            const float* __restrict__ b2, float* __restrict__ out) {
  extern __shared__ char smem[];
  half_t* xt  = (half_t*)(smem + SM_XT);   // [336][96]  halo tile (rows>=324 zero)
  half_t* ht  = (half_t*)(smem + SM_HT);   // [336][96]
  float*  rt  = (float*)(smem + SM_RT);    // [256][96]  conv2 result
  half_t* ws1 = (half_t*)(smem + SM_WS1);  // conv1 weights

  const int tx = blockIdx.x, ty = blockIdx.y, b = blockIdx.z;
  const int tid = threadIdx.x;
  const int lg = LANE >> 4, ln = LANE & 15;

  // warm L2 for the conv2 weight block (global_prefetch_b8)
  for (int off = tid * 128; off < 9 * O_ * O_ * 2; off += NT_ * 128)
    __builtin_prefetch((const char*)w2r + off, 0, 0);

  // stage conv1 weights to LDS
  for (int j = tid; j < (O_ * C_) / 8; j += NT_)
    ((uint4*)ws1)[j] = ((const uint4*)w1h)[j];

  // ---- stage halo (channel-major index -> coalesced pixel runs) ----
  for (int idx = tid; idx < 336 * C_; idx += NT_) {
    int c = idx / 336, p = idx % 336;
    half_t val = (half_t)0.f;
    if (p < 324) {
      int py = ty * 16 + (p / 18) - 1;
      int px = tx * 16 + (p % 18) - 1;
      py = py < 0 ? 0 : (py > H_ - 1 ? H_ - 1 : py);
      px = px < 0 ? 0 : (px > W_ - 1 ? W_ - 1 : px);
      val = (half_t)xattn[((b * C_ + c) * H_ + py) * W_ + px];
    }
    xt[p * C_ + c] = val;
  }
  __syncthreads();

  // ---- conv1 (1x1) + leaky: 126 tiles ----
  for (int tile = WAVE; tile < 21 * 6; tile += NWV_) {
    int mt = tile / 6, nt = tile % 6;
    v8f acc = {};
#pragma unroll
    for (int k = 0; k < 3; ++k) {
      v16h a  = ldA_h(xt, C_, mt * 16, k * 32, C_);
      v16h bm = ldBt_h(ws1, C_, k * 32, nt * 16, C_);
      acc = wmma_f16(a, bm, acc);
    }
    int c = nt * 16 + ln;
    float bb = b1[c];
#pragma unroll
    for (int v = 0; v < 8; ++v) {
      float z = acc[v] + bb;
      z = z >= 0.f ? z : 0.1f * z;
      ht[(mt * 16 + lg * 8 + v) * C_ + c] = (half_t)z;
    }
  }
  __syncthreads();

  // ---- conv2 (3x3) as 9 shifted GEMMs: 96 tiles, 27 WMMA each ----
  for (int tile = WAVE; tile < 16 * 6; tile += NWV_) {
    int mt = tile / 6, nt = tile % 6;
    v8f acc = {};
    int r  = mt * 16 + ln;
    int oy = r >> 4, ox = r & 15;
#pragma unroll
    for (int dy = 0; dy < 3; ++dy)
#pragma unroll
      for (int dx = 0; dx < 3; ++dx) {
        const half_t* arow = ht + ((oy + dy) * 18 + (ox + dx)) * C_;
        const half_t* wofs = w2r + (dy * 3 + dx) * O_ * O_;   // [o][i] for this tap
#pragma unroll
        for (int k = 0; k < 3; ++k) {
          v16h a;
#pragma unroll
          for (int i = 0; i < 16; ++i) a[i] = arow[k * 32 + a_kk(i, lg)];
          v16h bm = ldBt_h(wofs, O_, k * 32, nt * 16, O_);
          acc = wmma_f16(a, bm, acc);
        }
      }
    int c = nt * 16 + ln;
    float bb = b2[c];
#pragma unroll
    for (int v = 0; v < 8; ++v) {
      float z = acc[v] + bb;
      z = z >= 0.f ? z : 0.1f * z;
      rt[(mt * 16 + lg * 8 + v) * C_ + c] = z;
    }
  }
  __syncthreads();

  // ---- coalesced residual writeback: out = xattn + h2 ----
  for (int idx = tid; idx < 256 * C_; idx += NT_) {
    int c = idx / 256, p = idx % 256;
    int yy = ty * 16 + (p >> 4), xx = tx * 16 + (p & 15);
    int gi = ((b * C_ + c) * H_ + yy) * W_ + xx;
    out[gi] = xattn[gi] + rt[p * C_ + c];
  }
}

// ---------------------------------------------------------------------------
extern "C" void kernel_launch(void* const* d_in, const int* in_sizes, int n_in,
                              void* d_out, int out_size, void* d_ws, size_t ws_size,
                              hipStream_t stream) {
  const float* x    = (const float*)d_in[0];
  const float* bias = (const float*)d_in[1];
  const float* bpw  = (const float*)d_in[2];
  const float* qkvw = (const float*)d_in[3];
  const float* qkvb = (const float*)d_in[4];
  const float* outw = (const float*)d_in[5];
  const float* outb = (const float*)d_in[6];
  const float* w1   = (const float*)d_in[7];
  const float* b1   = (const float*)d_in[8];
  const float* w2   = (const float*)d_in[9];
  const float* b2   = (const float*)d_in[10];
  float* out = (float*)d_out;

  char* ws = (char*)d_ws;
  float*  abias = (float*)(ws + WS_ABIAS);
  half_t* qkvwh = (half_t*)(ws + WS_QKVW);
  half_t* outwh = (half_t*)(ws + WS_OUTW);
  half_t* w1h   = (half_t*)(ws + WS_W1);
  half_t* w2r   = (half_t*)(ws + WS_W2R);
  float*  xattn = (float*)(ws + WS_XATTN);

  prep_kernel<<<128, 256, 0, stream>>>(bias, bpw, qkvw, outw, w1, w2,
                                       abias, qkvwh, outwh, w1h, w2r);

  attn_kernel<<<B_ * NH_ * NH_, NT_, ATTN_SMEM, stream>>>(
      x, abias, qkvwh, outwh, qkvb, outb, xattn);

  conv_kernel<<<dim3(12, 12, 8), NT_, CONV_SMEM, stream>>>(
      xattn, w1h, w2r, b1, b2, out);
}
